// DynamicGraphLearner_42743514529960
// MI455X (gfx1250) — matrix operand order repaired and checked by reading, not compiled, not run on hardware
//
#include <hip/hip_runtime.h>
#include <hip/hip_bf16.h>
#include <math.h>

typedef __attribute__((ext_vector_type(16))) _Float16 v16h;
typedef __attribute__((ext_vector_type(8)))  float    v8f;

#define NODE_DIM  32
#define NUM_NODES 256
#define BATCH     32
#define HIDDEN    64
#define PAD       68   // row stride in floats; 16B-aligned, spreads rows across banks

// Fused DynamicGraphLearner, 32x32 output tile per block:
//   Ai = X @ W1[:, :32]^T + b1   (WMMA f16->f32, K=32 == NODE_DIM)
//   Aj = X @ W1[:, 32:]^T        (WMMA)
//   out[b,i,j] = sigmoid( sum_h relu(Ai[b,i,h] + Aj[b,j,h]) * W2[h] + b2 )
// 8 waves/block, 2 WMMA tiles per wave (one Ai tile, one Aj tile), then each
// thread produces 4 consecutive j outputs as one b128 store.

__device__ __forceinline__ v8f wmma_proj_tile(const float* __restrict__ X,
                                              const float* __restrict__ W1,
                                              int b, int rows0, int dofs,
                                              int htile, int lane)
{
    // A matrix: 16x32 f16, rows = node rows, K = d (ISA 7.12.2 16-bit A layout)
    // lane<16: halves t -> K in {0..7, 16..23}; lane>=16: K in {8..15, 24..31}
    const int m    = lane & 15;
    const int grow = rows0 + m;
    const float* xr = X + (size_t)(b * NUM_NODES + grow) * NODE_DIM;
    const int k0 = (lane & 16) ? 8 : 0;
    float4 x0 = *(const float4*)(xr + k0);
    float4 x1 = *(const float4*)(xr + k0 + 4);
    float4 x2 = *(const float4*)(xr + k0 + 16);
    float4 x3 = *(const float4*)(xr + k0 + 20);

    v16h amat;
    amat[0]  = (_Float16)x0.x; amat[1]  = (_Float16)x0.y;
    amat[2]  = (_Float16)x0.z; amat[3]  = (_Float16)x0.w;
    amat[4]  = (_Float16)x1.x; amat[5]  = (_Float16)x1.y;
    amat[6]  = (_Float16)x1.z; amat[7]  = (_Float16)x1.w;
    amat[8]  = (_Float16)x2.x; amat[9]  = (_Float16)x2.y;
    amat[10] = (_Float16)x2.z; amat[11] = (_Float16)x2.w;
    amat[12] = (_Float16)x3.x; amat[13] = (_Float16)x3.y;
    amat[14] = (_Float16)x3.z; amat[15] = (_Float16)x3.w;

    // B matrix: 32x16 f16, B[k, n] = W1[htile*16+n, dofs+k]
    // lane holds col n = lane&15; lane<16: K in {0..7, 16..23}; lane>=16: K in {8..15, 24..31}
    const int n = lane & 15;
    const int h = htile * 16 + n;
    const float* wr = W1 + h * (2 * NODE_DIM) + dofs + ((lane & 16) ? 8 : 0);
    float4 w0 = *(const float4*)(wr + 0);
    float4 w1 = *(const float4*)(wr + 4);
    float4 w2 = *(const float4*)(wr + 16);
    float4 w3 = *(const float4*)(wr + 20);

    v16h bmat;
    bmat[0]  = (_Float16)w0.x; bmat[1]  = (_Float16)w0.y;
    bmat[2]  = (_Float16)w0.z; bmat[3]  = (_Float16)w0.w;
    bmat[4]  = (_Float16)w1.x; bmat[5]  = (_Float16)w1.y;
    bmat[6]  = (_Float16)w1.z; bmat[7]  = (_Float16)w1.w;
    bmat[8]  = (_Float16)w2.x; bmat[9]  = (_Float16)w2.y;
    bmat[10] = (_Float16)w2.z; bmat[11] = (_Float16)w2.w;
    bmat[12] = (_Float16)w3.x; bmat[13] = (_Float16)w3.y;
    bmat[14] = (_Float16)w3.z; bmat[15] = (_Float16)w3.w;

    v8f c = {};
    c = __builtin_amdgcn_wmma_f32_16x16x32_f16(
        /*neg_a=*/false, amat, /*neg_b=*/false, bmat,
        /*c_mod=*/(short)0, c, /*reuse_a=*/false, /*reuse_b=*/false);
    return c;
}

__global__ __launch_bounds__(256)
void dgl_fused_kernel(const float* __restrict__ X,    // (B, N, 32)
                      const float* __restrict__ W1,   // (64, 64)
                      const float* __restrict__ b1,   // (64,)
                      const float* __restrict__ W2,   // (1, 64)
                      const float* __restrict__ b2,   // (1,)
                      float* __restrict__ out)        // (B, N, N)
{
    __shared__ __align__(16) float sAi[32][PAD];
    __shared__ __align__(16) float sAj[32][PAD];
    __shared__ __align__(16) float sW2[HIDDEN];

    const int tid  = threadIdx.x;
    const int wave = tid >> 5;          // 8 waves
    const int lane = tid & 31;          // wave32
    const int b    = blockIdx.z;
    const int i0   = blockIdx.y * 32;
    const int j0   = blockIdx.x * 32;

    if (tid < HIDDEN) sW2[tid] = W2[tid];

    // ---------------- stage 1: 2 WMMA tiles per wave (fully wave-uniform) ----
    const int rt    = wave >> 2;        // 16-row sub-tile: 0 or 1
    const int htile = wave & 3;         // h sub-tile: 0..3
    const int n     = lane & 15;
    const int h     = htile * 16 + n;
    const int rbase = rt * 16 + ((lane & 16) ? 8 : 0);

    v8f ci = wmma_proj_tile(X, W1, b, i0 + rt * 16, 0,        htile, lane); // Ai
    v8f cj = wmma_proj_tile(X, W1, b, j0 + rt * 16, NODE_DIM, htile, lane); // Aj

    const float bias = b1[h];
    #pragma unroll
    for (int r = 0; r < 8; ++r) sAi[rbase + r][h] = ci[r] + bias;
    #pragma unroll
    for (int r = 0; r < 8; ++r) sAj[rbase + r][h] = cj[r];

    __syncthreads();

    // ------------- stage 2: each thread -> 4 consecutive j outputs -----------
    const int i  = tid >> 3;            // 0..31
    const int jb = (tid & 7) * 4;       // 0,4,...,28
    const float4* ai4 = (const float4*)(&sAi[i][0]);
    const float4* aj0 = (const float4*)(&sAj[jb + 0][0]);
    const float4* aj1 = (const float4*)(&sAj[jb + 1][0]);
    const float4* aj2 = (const float4*)(&sAj[jb + 2][0]);
    const float4* aj3 = (const float4*)(&sAj[jb + 3][0]);
    const float4* w24 = (const float4*)(&sW2[0]);

    float acc0 = 0.0f, acc1 = 0.0f, acc2 = 0.0f, acc3 = 0.0f;
    #pragma unroll
    for (int q = 0; q < HIDDEN / 4; ++q) {
        const float4 ai = ai4[q];
        const float4 ww = w24[q];
        const float4 a0 = aj0[q];
        const float4 a1 = aj1[q];
        const float4 a2 = aj2[q];
        const float4 a3 = aj3[q];
        acc0 += fmaxf(ai.x + a0.x, 0.0f) * ww.x + fmaxf(ai.y + a0.y, 0.0f) * ww.y
              + fmaxf(ai.z + a0.z, 0.0f) * ww.z + fmaxf(ai.w + a0.w, 0.0f) * ww.w;
        acc1 += fmaxf(ai.x + a1.x, 0.0f) * ww.x + fmaxf(ai.y + a1.y, 0.0f) * ww.y
              + fmaxf(ai.z + a1.z, 0.0f) * ww.z + fmaxf(ai.w + a1.w, 0.0f) * ww.w;
        acc2 += fmaxf(ai.x + a2.x, 0.0f) * ww.x + fmaxf(ai.y + a2.y, 0.0f) * ww.y
              + fmaxf(ai.z + a2.z, 0.0f) * ww.z + fmaxf(ai.w + a2.w, 0.0f) * ww.w;
        acc3 += fmaxf(ai.x + a3.x, 0.0f) * ww.x + fmaxf(ai.y + a3.y, 0.0f) * ww.y
              + fmaxf(ai.z + a3.z, 0.0f) * ww.z + fmaxf(ai.w + a3.w, 0.0f) * ww.w;
    }

    const float bb = b2[0];
    float4 o;
    o.x = 1.0f / (1.0f + __expf(-(acc0 + bb)));
    o.y = 1.0f / (1.0f + __expf(-(acc1 + bb)));
    o.z = 1.0f / (1.0f + __expf(-(acc2 + bb)));
    o.w = 1.0f / (1.0f + __expf(-(acc3 + bb)));

    *(float4*)(&out[(size_t)(b * NUM_NODES + i0 + i) * NUM_NODES + (j0 + jb)]) = o;
}

extern "C" void kernel_launch(void* const* d_in, const int* in_sizes, int n_in,
                              void* d_out, int out_size, void* d_ws, size_t ws_size,
                              hipStream_t stream) {
    (void)in_sizes; (void)n_in; (void)d_ws; (void)ws_size; (void)out_size;
    const float* X  = (const float*)d_in[0];  // node_features (32,256,32)
    const float* W1 = (const float*)d_in[1];  // (64,64)
    const float* b1 = (const float*)d_in[2];  // (64,)
    const float* W2 = (const float*)d_in[3];  // (1,64)
    const float* b2 = (const float*)d_in[4];  // (1,)
    float* out = (float*)d_out;               // (32,256,256)

    dim3 grid(NUM_NODES / 32, NUM_NODES / 32, BATCH);
    dim3 block(256);
    dgl_fused_kernel<<<grid, block, 0, stream>>>(X, W1, b1, W2, b2, out);
}
